// GCNResNet_31920196944508
// MI455X (gfx1250) — matrix operand order repaired and checked by reading, not compiled, not run on hardware
//
#include <hip/hip_runtime.h>
#include <hip/hip_bf16.h>
#include <stdint.h>

// ---------------------------------------------------------------------------
// GCN-ResNet point-cloud net for MI455X (gfx1250, wave32, WMMA).
// Key insight: normalized adjacency == (adj + I)/21 exactly (row-degree is
// constant 21), so A@X is a 21-row gather, not a dense 2048x2048 bmm.
// Dense GEMMs use v_wmma_f32_16x16x32_bf16 (16x64 tile per wave).
// ---------------------------------------------------------------------------

#define BATCH 8
#define NPTS  2048
#define MROWS (BATCH * NPTS)   // 16384
#define KNN   20
#define BNEPS 1e-5f

typedef __attribute__((ext_vector_type(16))) __bf16 v16bf;
typedef __attribute__((ext_vector_type(8)))  float  v8f;

// ---------------- kNN: one block per (b, i); distance row in LDS ----------
__global__ void knn_kernel(const float* __restrict__ x, int* __restrict__ nbr) {
  __shared__ float dist[NPTS];
  __shared__ float rval[256];
  __shared__ int   ridx[256];
  const int b = blockIdx.x >> 11;          // / 2048
  const int i = blockIdx.x & (NPTS - 1);
  const float* xb = x + (size_t)b * NPTS * 3;
  const float xi0 = xb[i * 3 + 0], xi1 = xb[i * 3 + 1], xi2 = xb[i * 3 + 2];
  const int t = threadIdx.x;
  for (int j = t; j < NPTS; j += 256) {
    float a = xb[j * 3 + 0] - xi0;
    float c = xb[j * 3 + 1] - xi1;
    float d = xb[j * 3 + 2] - xi2;
    dist[j] = a * a + c * c + d * d;
  }
  __syncthreads();
  for (int s = 0; s < KNN; ++s) {
    float best = 3.4e38f; int bidx = NPTS;
    for (int j = t; j < NPTS; j += 256) {
      float v = dist[j];
      if (v < best || (v == best && j < bidx)) { best = v; bidx = j; }
    }
    rval[t] = best; ridx[t] = bidx;
    __syncthreads();
    for (int w = 128; w > 0; w >>= 1) {
      if (t < w) {
        float ov = rval[t + w]; int oi = ridx[t + w];
        if (ov < rval[t] || (ov == rval[t] && oi < ridx[t])) { rval[t] = ov; ridx[t] = oi; }
      }
      __syncthreads();
    }
    if (t == 0) {
      nbr[(size_t)blockIdx.x * KNN + s] = ridx[0];
      dist[ridx[0]] = 3.4e38f;
    }
    __syncthreads();
  }
}

// ---------------- gcn0 linear: (16384,3) @ (3,128) + b --------------------
__global__ void lin3_kernel(const float* __restrict__ x, const float* __restrict__ W,
                            const float* __restrict__ bias, float* __restrict__ out) {
  int idx = blockIdx.x * 256 + threadIdx.x;      // 16384*128 exact
  int m = idx >> 7, c = idx & 127;
  out[idx] = bias[c] + x[m * 3 + 0] * W[c] + x[m * 3 + 1] * W[128 + c]
                     + x[m * 3 + 2] * W[256 + c];
}

// ---------------- A-gather: out = relu((self + sum_knn)/21) --------------
__global__ void gather_relu_kernel(const float* __restrict__ lin, const int* __restrict__ nbr,
                                   float* __restrict__ out) {
  const int row = blockIdx.x;                 // b*2048 + i
  const int b = row >> 11;
  const int c = threadIdx.x;                  // C = 128
  const int* nb = nbr + (size_t)row * KNN;
  const size_t base = (size_t)b * NPTS * 128;
  float acc = lin[(size_t)row * 128 + c];     // the +I contribution
  #pragma unroll 4
  for (int k = 0; k < KNN; ++k) {
    acc += lin[base + (size_t)nb[k] * 128 + c];
  }
  out[(size_t)row * 128 + c] = fmaxf(acc * (1.0f / 21.0f), 0.0f);
}

// ---------------- f32 -> bf16 staging -------------------------------------
__global__ void cvt_bf16_kernel(const float* __restrict__ in, __hip_bfloat16* __restrict__ out,
                                int n) {
  int i = blockIdx.x * 256 + threadIdx.x;
  if (i < n) out[i] = __float2bfloat16(in[i]);
}

// ---------------- WMMA GEMM: Y[M,N] = Xb[M,K] @ Wb[K,N] + bias ------------
// Each wave computes a 16x64 slab: one A fragment reused across 4 WMMAs.
// A frag: lane%16 = M-row; dword v, lane-half h -> K = 16*(v/4)+8*h+2*(v%4).
// B frag: lane%16 = N; element e, lane-half h -> K = 8*h+e (e<8),
//         16+8*h+(e-8) (e>=8), matching A.
// D frag: lane%16 = N; dword v -> M = v + 8*(lane/16).
__global__ void gemm_bf16_wmma(const __hip_bfloat16* __restrict__ Xb,
                               const __hip_bfloat16* __restrict__ Wb,
                               const float* __restrict__ bias,
                               float* __restrict__ Y,
                               int M, int K, int N) {
  const int wave = (blockIdx.x * blockDim.x + threadIdx.x) >> 5;
  const int lane = threadIdx.x & 31;
  const int tilesN4 = N >> 6;                 // 64-wide slabs
  const int tM = wave / tilesN4;
  const int tN = wave % tilesN4;
  if (tM >= (M >> 4)) return;                 // wave-uniform; grid is exact anyway

  const int mrow = (tM << 4) + (lane & 15);
  const int half = lane >> 4;

  v8f acc[4];
  #pragma unroll
  for (int s = 0; s < 4; ++s) {
    const float bv = bias[(tN << 6) + (s << 4) + (lane & 15)];  // all 8 share this N
    #pragma unroll
    for (int v = 0; v < 8; ++v) acc[s][v] = bv;
  }

  const uint16_t* Xp = reinterpret_cast<const uint16_t*>(Xb);
  const uint16_t* Wp = reinterpret_cast<const uint16_t*>(Wb);

  for (int kb = 0; kb < K; kb += 32) {
    union { uint32_t u[8]; v16bf v; } a;
    const uint32_t* xrow = reinterpret_cast<const uint32_t*>(Xp + (size_t)mrow * K + kb);
    __builtin_prefetch(xrow + 16, 0, 1);      // global_prefetch next K-step
    #pragma unroll
    for (int v = 0; v < 8; ++v) {
      const int k0 = 16 * (v >> 2) + 8 * half + 2 * (v & 3);
      a.u[v] = xrow[k0 >> 1];
    }
    const uint16_t* wcol = Wp + (size_t)kb * N + (tN << 6) + (lane & 15);
    #pragma unroll
    for (int s = 0; s < 4; ++s) {
      union { uint16_t u[16]; v16bf v; } bfrag;
      #pragma unroll
      for (int e = 0; e < 16; ++e) {
        const int kk = (e < 8) ? (8 * half + e) : (16 + 8 * half + (e - 8));
        bfrag.u[e] = wcol[(size_t)kk * N + (s << 4)];
      }
      acc[s] = __builtin_amdgcn_wmma_f32_16x16x32_bf16(
          false, a.v, false, bfrag.v, (short)0, acc[s], false, false);
    }
  }

  #pragma unroll
  for (int s = 0; s < 4; ++s) {
    const int ncol = (tN << 6) + (s << 4) + (lane & 15);
    #pragma unroll
    for (int v = 0; v < 8; ++v) {
      const int m = (tM << 4) + v + 8 * half;
      Y[(size_t)m * N + ncol] = acc[s][v];
    }
  }
}

// ---------------- BatchNorm over (MROWS, C): stats then apply -------------
__global__ void bn_stats_kernel(const float* __restrict__ xin, float* __restrict__ stats,
                                int Mr, int C, int rowsPerBlock) {
  const int r0 = blockIdx.x * rowsPerBlock;
  const int r1 = min(r0 + rowsPerBlock, Mr);
  for (int c = threadIdx.x; c < C; c += blockDim.x) {
    float s = 0.f, s2 = 0.f;
    for (int r = r0; r < r1; ++r) {
      const float v = xin[(size_t)r * C + c];
      s += v; s2 += v * v;
    }
    atomicAdd(&stats[c], s);
    atomicAdd(&stats[C + c], s2);
  }
}

// flags: 1 = relu, 2 = add residual
__global__ void bn_apply_kernel(const float* __restrict__ y, const float* __restrict__ stats,
                                const float* __restrict__ g, const float* __restrict__ bb,
                                const float* __restrict__ residual, float* __restrict__ outb,
                                int Mr, int C, int flags) {
  const int idx = blockIdx.x * 256 + threadIdx.x;
  if (idx >= Mr * C) return;
  const int c = idx % C;
  const float inv = 1.0f / (float)Mr;
  const float mean = stats[c] * inv;
  const float var = stats[C + c] * inv - mean * mean;
  float v = (y[idx] - mean) * rsqrtf(var + BNEPS) * g[c] + bb[c];
  if (flags & 2) v += residual[idx];
  if (flags & 1) v = fmaxf(v, 0.f);
  outb[idx] = v;
}

// ---------------- mean over N: (8,2048,1024) -> (8,1024) ------------------
__global__ void pool_kernel(const float* __restrict__ h, float* __restrict__ pooled) {
  const int idx = blockIdx.x * 256 + threadIdx.x;  // 8192 exact
  const int b = idx >> 10, c = idx & 1023;
  const float* p = h + (size_t)b * NPTS * 1024 + c;
  float s = 0.f;
  for (int n = 0; n < NPTS; ++n) s += p[(size_t)n * 1024];
  pooled[idx] = s * (1.0f / (float)NPTS);
}

// ---------------- 8-row FC (+ optional BN-over-8 + relu) ------------------
__global__ void fc_bn_kernel(const float* __restrict__ in, const float* __restrict__ W,
                             const float* __restrict__ bias,
                             const float* __restrict__ g, const float* __restrict__ bb,
                             float* __restrict__ out, int K, int C, int do_bn) {
  __shared__ float sm[8 * 256];
  const int c = blockIdx.x;
  float acc[8];
  #pragma unroll
  for (int r = 0; r < 8; ++r) acc[r] = 0.f;
  for (int k = threadIdx.x; k < K; k += 256) {
    const float wv = W[(size_t)k * C + c];
    #pragma unroll
    for (int r = 0; r < 8; ++r) acc[r] += in[r * K + k] * wv;
  }
  #pragma unroll
  for (int r = 0; r < 8; ++r) sm[r * 256 + threadIdx.x] = acc[r];
  __syncthreads();
  for (int w = 128; w > 0; w >>= 1) {
    if (threadIdx.x < w) {
      #pragma unroll
      for (int r = 0; r < 8; ++r) sm[r * 256 + threadIdx.x] += sm[r * 256 + threadIdx.x + w];
    }
    __syncthreads();
  }
  if (threadIdx.x == 0) {
    float v[8]; float mean = 0.f;
    #pragma unroll
    for (int r = 0; r < 8; ++r) { v[r] = sm[r * 256] + bias[c]; mean += v[r]; }
    if (do_bn) {
      mean *= 0.125f;
      float var = 0.f;
      #pragma unroll
      for (int r = 0; r < 8; ++r) { const float d = v[r] - mean; var += d * d; }
      var *= 0.125f;
      const float sc = rsqrtf(var + BNEPS) * g[c];
      #pragma unroll
      for (int r = 0; r < 8; ++r) v[r] = fmaxf((v[r] - mean) * sc + bb[c], 0.f);
    }
    #pragma unroll
    for (int r = 0; r < 8; ++r) out[r * C + c] = v[r];
  }
}

// ---------------------------------------------------------------------------
extern "C" void kernel_launch(void* const* d_in, const int* in_sizes, int n_in,
                              void* d_out, int out_size, void* d_ws, size_t ws_size,
                              hipStream_t stream) {
  (void)in_sizes; (void)n_in; (void)out_size; (void)ws_size;
  auto F = [&](int i) { return (const float*)d_in[i]; };

  // input index map (setup_inputs dict order)
  const float* x = F(0);
  // 1 gcn0_W, 2 gcn0_b, 3 bn0_g, 4 bn0_b
  // 5+6*i: blk{W1,b1,W2,b2,g,b}
  // 23..26 l2, 27..30 l3, 31..34 l4, 35..38 l5, 39..42 l6, 43..44 l7

  // workspace layout (~123 MB)
  uint8_t* ws = (uint8_t*)d_ws;
  int*            nbr    = (int*)ws;                                  // 1.25 MB
  float*          feat   = (float*)(ws + 0x00140000);                 // 8 MB  (16384x128)
  float*          tmp0   = (float*)(ws + 0x00940000);                 // 64 MB (16384x1024)
  float*          tmp1   = (float*)(ws + 0x04940000);                 // 32 MB (16384x512)
  __hip_bfloat16* xb     = (__hip_bfloat16*)(ws + 0x06940000);        // 16 MB (16384x512 bf16)
  __hip_bfloat16* wb     = (__hip_bfloat16*)(ws + 0x07940000);        // 1 MB  (512x1024 bf16)
  float*          stats  = (float*)(ws + 0x07A40000);                 // 8 KB
  float*          pooled = (float*)(ws + 0x07A42000);                 // 8x1024
  float*          fc1    = (float*)(ws + 0x07A4A000);                 // 8x1024
  float*          fc2    = (float*)(ws + 0x07A52000);                 // 8x512

  const int M = MROWS;

  auto run_gemm = [&](const float* X, int K, const float* W, const float* bias,
                      int N, float* Y) {
    cvt_bf16_kernel<<<(M * K + 255) / 256, 256, 0, stream>>>(X, xb, M * K);
    cvt_bf16_kernel<<<(K * N + 255) / 256, 256, 0, stream>>>(W, wb, K * N);
    const int waves = (M / 16) * (N / 64);       // multiples of 8 for N in {128,512,1024}
    gemm_bf16_wmma<<<waves / 8, 256, 0, stream>>>(xb, wb, bias, Y, M, K, N);
  };
  auto run_bn = [&](const float* buf, int C, const float* g, const float* bbv,
                    const float* res, float* outb, int flags) {
    hipMemsetAsync(stats, 0, 2 * C * sizeof(float), stream);
    bn_stats_kernel<<<64, 256, 0, stream>>>(buf, stats, M, C, (M + 63) / 64);
    bn_apply_kernel<<<(M * C + 255) / 256, 256, 0, stream>>>(
        buf, stats, g, bbv, res ? res : buf, outb, M, C, flags);
  };

  // --- graph build: kNN adjacency as index lists (deg == 21 always) -------
  knn_kernel<<<MROWS, 256, 0, stream>>>(x, nbr);

  // --- gcn0 + bn0 ---------------------------------------------------------
  lin3_kernel<<<(M * 128) / 256, 256, 0, stream>>>(x, F(1), F(2), tmp0);
  gather_relu_kernel<<<MROWS, 128, 0, stream>>>(tmp0, nbr, feat);
  run_bn(feat, 128, F(3), F(4), nullptr, feat, 0);

  // --- 3 residual GCN blocks ---------------------------------------------
  for (int bi = 0; bi < 3; ++bi) {
    const int p = 5 + bi * 6;
    run_gemm(feat, 128, F(p + 0), F(p + 1), 128, tmp0);        // h@W1+b1
    gather_relu_kernel<<<MROWS, 128, 0, stream>>>(tmp0, nbr, tmp1);
    run_gemm(tmp1, 128, F(p + 2), F(p + 3), 128, tmp0);        // h1@W2+b2
    gather_relu_kernel<<<MROWS, 128, 0, stream>>>(tmp0, nbr, tmp1);
    run_bn(tmp1, 128, F(p + 4), F(p + 5), feat, feat, 3);      // relu(bn(h2)+h)
  }

  // --- per-point MLP ------------------------------------------------------
  run_gemm(feat, 128, F(23), F(24), 128, tmp0);
  run_bn(tmp0, 128, F(25), F(26), nullptr, tmp0, 1);
  run_gemm(tmp0, 128, F(27), F(28), 512, tmp1);
  run_bn(tmp1, 512, F(29), F(30), nullptr, tmp1, 1);
  run_gemm(tmp1, 512, F(31), F(32), 1024, tmp0);
  run_bn(tmp0, 1024, F(33), F(34), nullptr, tmp0, 1);

  // --- pool + head --------------------------------------------------------
  pool_kernel<<<(BATCH * 1024) / 256, 256, 0, stream>>>(tmp0, pooled);
  fc_bn_kernel<<<1024, 256, 0, stream>>>(pooled, F(35), F(36), F(37), F(38), fc1, 1024, 1024, 1);
  fc_bn_kernel<<<512,  256, 0, stream>>>(fc1,    F(39), F(40), F(41), F(42), fc2, 1024, 512, 1);
  fc_bn_kernel<<<40,   256, 0, stream>>>(fc2,    F(43), F(44), nullptr, nullptr,
                                         (float*)d_out, 512, 40, 0);
}
